// multi_head_attn_17282948399586
// MI455X (gfx1250) — compile-verified, matrix operand order checked by reading, not run on hardware
//
#include <hip/hip_runtime.h>
#include <hip/hip_bf16.h>
#include <math.h>

#define DMODEL 1024
#define HEADS  16
#define DK     64
#define SEQ    2048
#define BATCH  2

typedef __attribute__((ext_vector_type(16))) __bf16 v16bf;
typedef __attribute__((ext_vector_type(4)))  __bf16 v4bf;
typedef __attribute__((ext_vector_type(8)))  float  v8f;

// ds_swizzle XOR butterfly (group-of-32 mode: and=0x1f, or=0, xor=m)
#define SWZ(v, m) __builtin_bit_cast(float, \
    __builtin_amdgcn_ds_swizzle(__builtin_bit_cast(int, (v)), (((m) << 10) | 0x1f)))

// Async 16-byte global -> LDS copy (per lane), tracked with ASYNCcnt.
__device__ __forceinline__ void async_load16(unsigned lds_off, const void* gptr) {
  asm volatile("global_load_async_to_lds_b128 %0, %1, off"
               :: "v"(lds_off), "v"((unsigned long long)(uintptr_t)gptr)
               : "memory");
}
__device__ __forceinline__ void wait_async0() {
  asm volatile("s_wait_asynccnt 0x0" ::: "memory");
}

// 16x32 (A) or 32x16-as-K-major (B) bf16 fragment load, per CDNA5 ISA layout:
// line r = lane&15; two 8-element K chunks at (lane>>4)*8 and +16.
// Works for global and LDS pointers.
__device__ __forceinline__ v16bf load_frag(const __bf16* base, int ld) {
  const int lane = threadIdx.x & 31;
  const int r    = lane & 15;
  const int kc   = (lane >> 4) << 3;          // 0 or 8
  const __bf16* p = base + (size_t)r * ld + kc;
  union { v16bf v; uint4 u[2]; } f;
  f.u[0] = *reinterpret_cast<const uint4*>(p);
  f.u[1] = *reinterpret_cast<const uint4*>(p + 16);
  return f.v;
}

__global__ void cvt_f32_bf16_k(const float* __restrict__ x, __bf16* __restrict__ y, int n) {
  int i = (blockIdx.x * blockDim.x + threadIdx.x) * 4;
  if (i < n) {
    float4 f = *reinterpret_cast<const float4*>(x + i);
    v4bf o;
    o[0] = (__bf16)f.x; o[1] = (__bf16)f.y; o[2] = (__bf16)f.z; o[3] = (__bf16)f.w;
    *reinterpret_cast<v4bf*>(y + i) = o;
  }
}

// Pack int32 mask [S,S] into bitmask [S, S/32] (bit=1 means keep score).
__global__ void pack_mask_k(const int* __restrict__ mask, unsigned* __restrict__ bits, int nw) {
  int w = blockIdx.x * blockDim.x + threadIdx.x;
  if (w >= nw) return;
  const int* p = mask + (size_t)w * 32;
  unsigned v = 0;
#pragma unroll
  for (int i = 0; i < 32; ++i) v |= (p[i] != 0 ? 1u : 0u) << i;
  bits[w] = v;
}

// Y = X @ W^T + bias.  X:[M,K] bf16, W:[N,K] bf16 (both K-major).
// Block: 256 threads = 8 waves in 4(M) x 2(N); macro-tile 64M x 128N.
// A/B panels double-buffered in LDS via async-to-LDS.
// mode 0: bf16 [B,H,S,DK] | mode 1: bf16 [B,H,DK,S] | mode 2: fp32 [M,N]
__global__ void gemm_wmma_k(const __bf16* __restrict__ X, const __bf16* __restrict__ W,
                            const float* __restrict__ bias, void* __restrict__ Y,
                            int M, int N, int K, int mode) {
  __shared__ __align__(16) __bf16 smA[2][64 * 32];    // 2 x 4 KB
  __shared__ __align__(16) __bf16 smB[2][128 * 32];   // 2 x 8 KB

  const int tid  = threadIdx.x;
  const int wave = tid >> 5;
  const int lane = tid & 31;
  const int nblk = N >> 7;                 // N / 128
  const int bm   = blockIdx.x / nblk;
  const int bn   = blockIdx.x % nblk;
  const int m0   = bm * 64;
  const int n0   = bn * 128;
  const int wm   = wave >> 1;              // 0..3  (16-row strip)
  const int wn   = wave & 1;               // 0..1  (64-col strip)

  v8f acc[4];
  const v8f vz = {};
#pragma unroll
  for (int dt = 0; dt < 4; ++dt) acc[dt] = vz;

  auto stage = [&](int buf, int k0) {
    {   // A panel: 64 rows x 32 k  = 256 x 16B chunks, one per thread
      const int row = tid >> 2, c = (tid & 3) * 8;
      async_load16((unsigned)(uintptr_t)&smA[buf][row * 32 + c],
                   X + (size_t)(m0 + row) * K + k0 + c);
    }
#pragma unroll
    for (int i = 0; i < 2; ++i) {   // B panel: 128 rows x 32 k = 512 chunks
      const int cid = tid + i * 256;
      const int row = cid >> 2, c = (cid & 3) * 8;
      async_load16((unsigned)(uintptr_t)&smB[buf][row * 32 + c],
                   W + (size_t)(n0 + row) * K + k0 + c);
    }
  };

  stage(0, 0);
  wait_async0();
  __syncthreads();

  int buf = 0;
  for (int k0 = 0; k0 < K; k0 += 32) {
    if (k0 + 32 < K) stage(buf ^ 1, k0 + 32);
    v16bf a = load_frag(&smA[buf][(wm * 16) * 32], 32);
#pragma unroll
    for (int dt = 0; dt < 4; ++dt) {
      v16bf b = load_frag(&smB[buf][(wn * 64 + dt * 16) * 32], 32);
      acc[dt] = __builtin_amdgcn_wmma_f32_16x16x32_bf16(false, a, false, b, (short)0,
                                                        acc[dt], false, false);
    }
    wait_async0();        // my prefetches for buf^1 have landed
    __syncthreads();      // everyone done reading buf; everyone's buf^1 visible
    buf ^= 1;
  }

  const int n  = lane & 15;
  const int mh = (lane >> 4) << 3;
#pragma unroll
  for (int dt = 0; dt < 4; ++dt) {
    const int col = n0 + wn * 64 + dt * 16 + n;
    const float bv = bias[col];
#pragma unroll
    for (int r = 0; r < 8; ++r) {
      const int row = m0 + wm * 16 + mh + r;
      const float v = acc[dt][r] + bv;
      if (mode == 0) {
        int b_ = row >> 11, s = row & (SEQ - 1), h = col >> 6, d = col & (DK - 1);
        ((__bf16*)Y)[(((size_t)(b_ * HEADS + h)) * SEQ + s) * DK + d] = (__bf16)v;
      } else if (mode == 1) {
        int b_ = row >> 11, s = row & (SEQ - 1), h = col >> 6, d = col & (DK - 1);
        ((__bf16*)Y)[(((size_t)(b_ * HEADS + h)) * DK + d) * SEQ + s] = (__bf16)v;
      } else {
        ((float*)Y)[(size_t)row * N + col] = v;
      }
    }
  }
}

// Flash attention: 256 threads = 8 waves, each wave owns 16 query rows of one (b,h).
// K (32x64) and V (64x32) tiles are async-staged into LDS, double-buffered, and
// shared by all 8 waves.  Qh,Kh: [B*H,S,DK] bf16.  Vt: [B*H,DK,S] bf16.
// mbits: [S, S/32].  ctx: [B, S, H*DK] bf16.
__global__ void attn_flash_k(const __bf16* __restrict__ Qh, const __bf16* __restrict__ Kh,
                             const __bf16* __restrict__ Vt,
                             const unsigned* __restrict__ mbits,
                             __bf16* __restrict__ ctx) {
  __shared__ __align__(16) __bf16 smK[2][32 * 64];   // 2 x 4 KB
  __shared__ __align__(16) __bf16 smV[2][64 * 32];   // 2 x 4 KB
  __shared__ __align__(16) __bf16 pst[8][16 * 32];   // 8 KB per-wave P staging

  const int tid  = threadIdx.x;
  const int wave = tid >> 5;
  const int lane = tid & 31;
  const int bh   = blockIdx.y;
  const int b    = bh >> 4;
  const int h    = bh & (HEADS - 1);
  const int q0   = (blockIdx.x * 8 + wave) * 16;

  const __bf16* Qbase = Qh + ((size_t)bh * SEQ + q0) * DK;
  const v16bf qa0 = load_frag(Qbase,      DK);
  const v16bf qa1 = load_frag(Qbase + 32, DK);

  const int n  = lane & 15;
  const int mh = (lane >> 4) << 3;

  float run_max[8], run_sum[8];
  v8f acc[4];
  const v8f vz = {};
#pragma unroll
  for (int r = 0; r < 8; ++r) { run_max[r] = -__builtin_inff(); run_sum[r] = 0.f; }
#pragma unroll
  for (int dt = 0; dt < 4; ++dt) acc[dt] = vz;

  __bf16* ps = &pst[wave][0];

  auto stageKV = [&](int buf, int k0) {
    {   // K tile: 32 key rows x 64 d = 256 x 16B chunks, one per thread
      const int row = tid >> 3, cc = (tid & 7) * 8;
      async_load16((unsigned)(uintptr_t)&smK[buf][row * 64 + cc],
                   Kh + ((size_t)bh * SEQ + k0 + row) * DK + cc);
    }
    {   // V tile: 64 d rows x 32 s = 256 x 16B chunks, one per thread
      const int row = tid >> 2, cc = (tid & 3) * 8;
      async_load16((unsigned)(uintptr_t)&smV[buf][row * 32 + cc],
                   Vt + ((size_t)bh * DK + row) * SEQ + k0 + cc);
    }
  };

  stageKV(0, 0);
  wait_async0();
  __syncthreads();

  int buf = 0;
  for (int k0 = 0; k0 < SEQ; k0 += 32) {
    if (k0 + 32 < SEQ) stageKV(buf ^ 1, k0 + 32);

    // one mask word per row covers this whole 32-key step
    unsigned mword[8];
#pragma unroll
    for (int r = 0; r < 8; ++r)
      mword[r] = mbits[((size_t)(q0 + mh + r) << 6) + (k0 >> 5)];

    float sv[2][8];
#pragma unroll
    for (int t = 0; t < 2; ++t) {
      v16bf kb0 = load_frag(&smK[buf][(t * 16) * 64],      64);
      v16bf kb1 = load_frag(&smK[buf][(t * 16) * 64 + 32], 64);
      v8f s = {};
      s = __builtin_amdgcn_wmma_f32_16x16x32_bf16(false, qa0, false, kb0, (short)0, s,
                                                  false, false);
      s = __builtin_amdgcn_wmma_f32_16x16x32_bf16(false, qa1, false, kb1, (short)0, s,
                                                  false, false);
      const unsigned bit = 1u << (t * 16 + n);
#pragma unroll
      for (int r = 0; r < 8; ++r) {
        float x = s[r] * 0.125f;                       // 1/sqrt(DK)
        if ((mword[r] & bit) == 0) x = 1e-9f;          // reference mask quirk
        sv[t][r] = x;
      }
    }
    // online softmax (reduce over 16 lanes of each half-wave)
#pragma unroll
    for (int r = 0; r < 8; ++r) {
      float m2 = fmaxf(sv[0][r], sv[1][r]);
      m2 = fmaxf(m2, SWZ(m2, 1));
      m2 = fmaxf(m2, SWZ(m2, 2));
      m2 = fmaxf(m2, SWZ(m2, 4));
      m2 = fmaxf(m2, SWZ(m2, 8));
      const float nm   = fmaxf(run_max[r], m2);
      const float corr = __expf(run_max[r] - nm);
      const float e0 = __expf(sv[0][r] - nm);
      const float e1 = __expf(sv[1][r] - nm);
      ps[(mh + r) * 32 + n]      = (__bf16)e0;
      ps[(mh + r) * 32 + 16 + n] = (__bf16)e1;
      float psum = e0 + e1;
      psum += SWZ(psum, 1);
      psum += SWZ(psum, 2);
      psum += SWZ(psum, 4);
      psum += SWZ(psum, 8);
      run_sum[r] = run_sum[r] * corr + psum;
      run_max[r] = nm;
#pragma unroll
      for (int dt = 0; dt < 4; ++dt) acc[dt][r] *= corr;
    }
    // P (16x32) @ V (32x64) accumulate, V from LDS
    const v16bf pa = load_frag(ps, 32);
#pragma unroll
    for (int dt = 0; dt < 4; ++dt) {
      v16bf vb = load_frag(&smV[buf][(dt * 16) * 32], 32);
      acc[dt] = __builtin_amdgcn_wmma_f32_16x16x32_bf16(false, pa, false, vb, (short)0,
                                                        acc[dt], false, false);
    }
    wait_async0();
    __syncthreads();
    buf ^= 1;
  }

#pragma unroll
  for (int r = 0; r < 8; ++r) {
    const float inv = 1.0f / run_sum[r];
    const int row = q0 + mh + r;
#pragma unroll
    for (int dt = 0; dt < 4; ++dt) {
      const int col = h * DK + dt * 16 + n;
      ctx[((size_t)b * SEQ + row) * DMODEL + col] = (__bf16)(acc[dt][r] * inv);
    }
  }
}

extern "C" void kernel_launch(void* const* d_in, const int* in_sizes, int n_in,
                              void* d_out, int out_size, void* d_ws, size_t ws_size,
                              hipStream_t stream) {
  const float* q    = (const float*)d_in[0];
  const float* k    = (const float*)d_in[1];
  const float* v    = (const float*)d_in[2];
  const int*   mask = (const int*)  d_in[3];
  const float* wq_w = (const float*)d_in[4];
  const float* wq_b = (const float*)d_in[5];
  const float* wk_w = (const float*)d_in[6];
  const float* wk_b = (const float*)d_in[7];
  const float* wv_w = (const float*)d_in[8];
  const float* wv_b = (const float*)d_in[9];
  const float* wo_w = (const float*)d_in[10];
  const float* wo_b = (const float*)d_in[11];

  const int NTOK  = BATCH * SEQ;            // 4096
  const int NACT  = NTOK * DMODEL;          // 4,194,304
  const int NWGT  = DMODEL * DMODEL;        // 1,048,576
  const int NMW   = SEQ * (SEQ / 32);       // 131,072 mask words

  char* ws = (char*)d_ws;
  size_t off = 0;
  auto carve = [&](size_t bytes) { void* p = ws + off; off += bytes; return p; };

  __bf16* qb    = (__bf16*)carve((size_t)NACT * 2);
  __bf16* kb    = (__bf16*)carve((size_t)NACT * 2);
  __bf16* vb    = (__bf16*)carve((size_t)NACT * 2);
  __bf16* wqb   = (__bf16*)carve((size_t)NWGT * 2);
  __bf16* wkb   = (__bf16*)carve((size_t)NWGT * 2);
  __bf16* wvb   = (__bf16*)carve((size_t)NWGT * 2);
  __bf16* wob   = (__bf16*)carve((size_t)NWGT * 2);
  __bf16* Qh    = (__bf16*)carve((size_t)NACT * 2);
  __bf16* Kh    = (__bf16*)carve((size_t)NACT * 2);
  __bf16* Vt    = (__bf16*)carve((size_t)NACT * 2);
  __bf16* ctx   = (__bf16*)carve((size_t)NACT * 2);
  unsigned* mbits = (unsigned*)carve((size_t)NMW * 4);
  (void)ws_size; (void)in_sizes; (void)n_in; (void)out_size;

  // 1) fp32 -> bf16 conversions (4 elems/thread) + mask bit-pack
  cvt_f32_bf16_k<<<NACT / 4 / 256, 256, 0, stream>>>(q, qb, NACT);
  cvt_f32_bf16_k<<<NACT / 4 / 256, 256, 0, stream>>>(k, kb, NACT);
  cvt_f32_bf16_k<<<NACT / 4 / 256, 256, 0, stream>>>(v, vb, NACT);
  cvt_f32_bf16_k<<<NWGT / 4 / 256, 256, 0, stream>>>(wq_w, wqb, NWGT);
  cvt_f32_bf16_k<<<NWGT / 4 / 256, 256, 0, stream>>>(wk_w, wkb, NWGT);
  cvt_f32_bf16_k<<<NWGT / 4 / 256, 256, 0, stream>>>(wv_w, wvb, NWGT);
  cvt_f32_bf16_k<<<NWGT / 4 / 256, 256, 0, stream>>>(wo_w, wob, NWGT);
  pack_mask_k<<<NMW / 256, 256, 0, stream>>>(mask, mbits, NMW);

  // 2) projections: 64x128 macro-tiles, async double-buffered LDS staging
  const int gemmBlocks = (NTOK / 64) * (DMODEL / 128);   // 512
  gemm_wmma_k<<<gemmBlocks, 256, 0, stream>>>(qb, wqb, wq_b, Qh, NTOK, DMODEL, DMODEL, 0);
  gemm_wmma_k<<<gemmBlocks, 256, 0, stream>>>(kb, wkb, wk_b, Kh, NTOK, DMODEL, DMODEL, 0);
  gemm_wmma_k<<<gemmBlocks, 256, 0, stream>>>(vb, wvb, wv_b, Vt, NTOK, DMODEL, DMODEL, 1);

  // 3) flash attention: grid (S/128, B*H), 8 waves/block, 16 q-rows per wave,
  //    K/V tiles async-staged in LDS and shared by all waves
  attn_flash_k<<<dim3(SEQ / 128, BATCH * HEADS), 256, 0, stream>>>(Qh, Kh, Vt, mbits, ctx);

  // 4) output projection -> fp32 d_out
  gemm_wmma_k<<<gemmBlocks, 256, 0, stream>>>(ctx, wob, wo_b, (float*)d_out,
                                              NTOK, DMODEL, DMODEL, 2);
}
